// ConvlstmTransBlock_34737695490071
// MI455X (gfx1250) — matrix operand order repaired and checked by reading, compiled, not run on hardware
//
#include <hip/hip_runtime.h>
#include <hip/hip_bf16.h>

typedef __attribute__((ext_vector_type(16))) __bf16 v16bf;
typedef __attribute__((ext_vector_type(8)))  float  v8f;

#define B_    4
#define T_    16
#define H_    48
#define W_    48
#define CIN   64
#define F_    64
#define HU    96
#define WU    96
#define CMAP  32
#define COUT  256              // 4*F
#define KW    576              // 3*3*CIN
#define NPIX  (B_*HU*WU)       // 36864 pixels per step
#define ALPHA 0.3f
#define BNEPS 1e-3f

// ---------------- prep kernels ----------------

// Convert + transpose weights: [3,3,Cin,256] fp32 -> [256][576] bf16
__global__ void prep_weights(const float* __restrict__ Wx,
                             const float* __restrict__ Wh,
                             __bf16* __restrict__ wxt,
                             __bf16* __restrict__ wht) {
    int i = blockIdx.x * blockDim.x + threadIdx.x;   // 2*256*576 threads
    int conv = i / (COUT * KW);
    int r    = i % (COUT * KW);
    int n = r / KW;
    int k = r % KW;
    const float* src = conv ? Wh : Wx;
    __bf16 v = (__bf16)src[(size_t)k * COUT + n];
    (conv ? wht : wxt)[(size_t)n * KW + k] = v;
}

__global__ void prep_x(const float* __restrict__ x, __bf16* __restrict__ xbf) {
    int i = blockIdx.x * blockDim.x + threadIdx.x;   // B*T*H*W*CIN threads
    xbf[i] = (__bf16)x[i];
}

__global__ void prep_state(const float* __restrict__ h0,
                           const float* __restrict__ c0,
                           __bf16* __restrict__ hbf,
                           float* __restrict__ c) {
    int i = blockIdx.x * blockDim.x + threadIdx.x;   // NPIX*F threads
    hbf[i] = (__bf16)h0[i];
    c[i]   = c0[i];
}

// Broadcast skip map into output channels [0,32) for all t
__global__ void write_skip(const float* __restrict__ map_, float* __restrict__ out) {
    int i = blockIdx.x * blockDim.x + threadIdx.x;   // B*T*HU*WU*CMAP threads
    int cm   = i % CMAP;
    int rest = i / CMAP;
    int yx   = rest % (HU * WU);
    int bt   = rest / (HU * WU);
    int t = bt % T_;
    int b = bt / T_;
    out[((size_t)(b * T_ + t) * (HU * WU) + yx) * (CMAP + F_) + cm] =
        map_[((size_t)b * (HU * WU) + yx) * CMAP + cm];
}

// ---------------- implicit-GEMM ConvLSTM gate pre-activation ----------------
// z[p][0:256] = conv(x_up_t, Wx)[p] + conv(h_prev, Wh)[p] + bias
// Block: 32-pixel strip (along Wu) x all 256 channels. 4 waves; wave w owns
// N in [64w, 64w+64). Each wave computes a 32(M) x 64(N) tile: two 16-row A
// fragments share every B (weight) fragment -> 8 WMMAs per 12 b128 loads.
__global__ __launch_bounds__(128)
void conv_step(const __bf16* __restrict__ xbf,
               const __bf16* __restrict__ hbf,
               const __bf16* __restrict__ wxt,
               const __bf16* __restrict__ wht,
               const float* __restrict__ bias,
               float* __restrict__ z, int t) {
    const int strip = blockIdx.x;            // NPIX/32 strips
    const int lane  = threadIdx.x & 31;
    const int wave  = threadIdx.x >> 5;
    const int P0  = strip * 32;
    const int b   = P0 / (HU * WU);
    const int rem = P0 % (HU * WU);
    const int yu  = rem / WU;
    const int xu0 = rem % WU;                // WU%32==0 -> strip stays in one row
    const int ml   = lane & 15;              // A row (within 16-row fragment)
    const int kbA  = (lane < 16) ? 0 : 8;    // A K-split per ISA 16-bit A layout
    const int kbB  = (lane < 16) ? 0 : 8;    // B K-split matching A (table order)
    const int ncol = lane & 15;
    const int nbase = wave * 64;

    union Frag { v16bf v; uint4 q[2]; };

    v8f acc[2][4];
#pragma unroll
    for (int j = 0; j < 4; ++j) {
        float bv = bias[nbase + j * 16 + ncol];
#pragma unroll
        for (int r = 0; r < 8; ++r) { acc[0][j][r] = bv; acc[1][j][r] = bv; }
    }

#pragma unroll
    for (int conv = 0; conv < 2; ++conv) {
        const __bf16* wt = conv ? wht : wxt;
#pragma unroll
        for (int tap = 0; tap < 9; ++tap) {
            const int ky = tap / 3, kx = tap % 3;
            const int ys  = yu + ky - 1;                // uniform over wave
            const int xs0 = xu0 + ml + kx - 1;          // rows 0..15
            const int xs1 = xs0 + 16;                   // rows 16..31
            const bool vy = (ys >= 0) && (ys < HU);
            const bool v0 = vy && (xs0 >= 0) && (xs0 < WU);
            const bool v1 = vy && (xs1 >= 0) && (xs1 < WU);
            const __bf16 *src0, *src1;
            if (conv == 0) {   // upsampled x: nearest-neighbor fold into address
                const __bf16* base =
                    xbf + (((size_t)b * T_ + t) * H_ + (ys >> 1)) * W_ * CIN;
                src0 = base + (size_t)(xs0 >> 1) * CIN;
                src1 = base + (size_t)(xs1 >> 1) * CIN;
            } else {
                const __bf16* base = hbf + ((size_t)b * HU + ys) * WU * F_;
                src0 = base + (size_t)xs0 * F_;
                src1 = base + (size_t)xs1 * F_;
            }
#pragma unroll
            for (int cc = 0; cc < 2; ++cc) {            // two 32-channel K chunks
                Frag a0, a1;
                if (v0) {
                    const __bf16* p = src0 + cc * 32 + kbA;
                    a0.q[0] = *(const uint4*)(p);
                    a0.q[1] = *(const uint4*)(p + 16);
                } else {
                    a0.q[0] = make_uint4(0u, 0u, 0u, 0u);
                    a0.q[1] = make_uint4(0u, 0u, 0u, 0u);
                }
                if (v1) {
                    const __bf16* p = src1 + cc * 32 + kbA;
                    a1.q[0] = *(const uint4*)(p);
                    a1.q[1] = *(const uint4*)(p + 16);
                } else {
                    a1.q[0] = make_uint4(0u, 0u, 0u, 0u);
                    a1.q[1] = make_uint4(0u, 0u, 0u, 0u);
                }
                const int koff = tap * CIN + cc * 32 + kbB;
#pragma unroll
                for (int j = 0; j < 4; ++j) {
                    Frag bw;
                    const __bf16* wp = wt + (size_t)(nbase + j * 16 + ncol) * KW + koff;
                    bw.q[0] = *(const uint4*)(wp);
                    bw.q[1] = *(const uint4*)(wp + 16);
                    acc[0][j] = __builtin_amdgcn_wmma_f32_16x16x32_bf16(
                        false, a0.v, false, bw.v, (short)0, acc[0][j], false, false);
                    acc[1][j] = __builtin_amdgcn_wmma_f32_16x16x32_bf16(
                        false, a1.v, false, bw.v, (short)0, acc[1][j], false, false);
                }
            }
        }
    }

    // C/D layout: VGPR r, lanes 0-15 -> M=r, lanes 16-31 -> M=8+r; N = lane&15
    const int mrow = (lane >> 4) << 3;
#pragma unroll
    for (int mt = 0; mt < 2; ++mt) {
#pragma unroll
        for (int j = 0; j < 4; ++j) {
            const int n = nbase + j * 16 + ncol;
#pragma unroll
            for (int r = 0; r < 8; ++r) {
                z[(size_t)(P0 + mt * 16 + mrow + r) * COUT + n] = acc[mt][j][r];
            }
        }
    }
}

// ---------------- gates + state update + BN-fused output ----------------
__global__ void gate_update(const float* __restrict__ z,
                            float* __restrict__ c,
                            __bf16* __restrict__ hbf,
                            float* __restrict__ out,
                            const float* __restrict__ gamma,
                            const float* __restrict__ beta,
                            const float* __restrict__ mmean,
                            const float* __restrict__ mvar,
                            int t) {
    int idx = blockIdx.x * blockDim.x + threadIdx.x;   // NPIX*F threads
    int p = idx >> 6;
    int f = idx & 63;
    const float* zp = z + (size_t)p * COUT;
    float zi = zp[f], zf = zp[64 + f], zg = zp[128 + f], zo = zp[192 + f];
    float hs_i = fminf(fmaxf(0.2f * zi + 0.5f, 0.0f), 1.0f);
    float hs_f = fminf(fmaxf(0.2f * zf + 0.5f, 0.0f), 1.0f);
    float hs_o = fminf(fmaxf(0.2f * zo + 0.5f, 0.0f), 1.0f);
    float lg = (zg >= 0.0f) ? zg : ALPHA * zg;
    float cn = hs_f * c[idx] + hs_i * lg;
    float lc = (cn >= 0.0f) ? cn : ALPHA * cn;
    float hn = hs_o * lc;
    c[idx]   = cn;
    hbf[idx] = (__bf16)hn;                    // recurrent input for next step
    int b  = p / (HU * WU);
    int yx = p % (HU * WU);
    float s = gamma[f] * rsqrtf(mvar[f] + BNEPS);
    out[((size_t)(b * T_ + t) * (HU * WU) + yx) * (CMAP + F_) + CMAP + f] =
        (hn - mmean[f]) * s + beta[f];
}

// ---------------- launch ----------------
extern "C" void kernel_launch(void* const* d_in, const int* in_sizes, int n_in,
                              void* d_out, int out_size, void* d_ws, size_t ws_size,
                              hipStream_t stream) {
    const float* inputs = (const float*)d_in[0];
    const float* h0     = (const float*)d_in[1];
    const float* c0     = (const float*)d_in[2];
    const float* map_   = (const float*)d_in[3];
    const float* Wx     = (const float*)d_in[4];
    const float* Wh     = (const float*)d_in[5];
    const float* bias   = (const float*)d_in[6];
    const float* gamma  = (const float*)d_in[7];
    const float* beta   = (const float*)d_in[8];
    const float* mmean  = (const float*)d_in[9];
    const float* mvar   = (const float*)d_in[10];
    float* out = (float*)d_out;

    char* ws = (char*)d_ws;
    float*  zbuf = (float*)(ws + 0);                 // 37,748,736 B
    float*  cbuf = (float*)(ws + 37748736);          //  9,437,184 B
    __bf16* hbf  = (__bf16*)(ws + 47185920);         //  4,718,592 B
    __bf16* wxt  = (__bf16*)(ws + 51904512);         //    294,912 B
    __bf16* wht  = (__bf16*)(ws + 52199424);         //    294,912 B
    __bf16* xbf  = (__bf16*)(ws + 52494336);         // 18,874,368 B (total ~68 MB)

    prep_weights<<<(2 * COUT * KW) / 256, 256, 0, stream>>>(Wx, Wh, wxt, wht);
    prep_x<<<(B_ * T_ * H_ * W_ * CIN) / 256, 256, 0, stream>>>(inputs, xbf);
    prep_state<<<(NPIX * F_) / 256, 256, 0, stream>>>(h0, c0, hbf, cbuf);
    write_skip<<<(B_ * T_ * HU * WU * CMAP) / 256, 256, 0, stream>>>(map_, out);

    for (int t = 0; t < T_; ++t) {
        conv_step<<<NPIX / 32, 128, 0, stream>>>(xbf, hbf, wxt, wht, bias, zbuf, t);
        gate_update<<<(NPIX * F_) / 256, 256, 0, stream>>>(
            zbuf, cbuf, hbf, out, gamma, beta, mmean, mvar, t);
    }
}
